// ResSwinTransformerGroup_2826088480968
// MI455X (gfx1250) — matrix operand order repaired and checked by reading, not compiled, not run on hardware
//
#include <hip/hip_runtime.h>
#include <hip/hip_bf16.h>
#include <stdint.h>

typedef _Float16 half_t;
typedef __attribute__((ext_vector_type(16))) _Float16 v16h;
typedef __attribute__((ext_vector_type(8)))  float    v8f;

#define DIM_   192
#define HEADS_ 6
#define HW_    4096
#define MROWS_ 65536   // B*H*W = 16*4096
#define SHIFT_ 4
static constexpr float ATT_SCALE = 0.17677669529663687f; // 32^-0.5

union FragH { v16h v; half_t h[16]; uint32_t u[8]; };
union FragF { v8f  v; float  f[8]; };

// ---------------- WMMA fragment loaders (wave32, 16x16x32 f16) ----------------
// A (16x32 f16): lane = (row%16) + 16*half; VGPR0..7 hold K = half*16 + {0..15}
// (pairs packed per dword).
__device__ inline v16h frag_a_rm(const half_t* A, int lda, int row0, int kb) {
  int lane = threadIdx.x & 31;
  const half_t* p = A + (size_t)(row0 + (lane & 15)) * lda + kb + ((lane >> 4) << 4);
  const uint32_t* p0 = (const uint32_t*)p;
  const uint32_t* p1 = (const uint32_t*)(p + 8);
  FragH r;
#pragma unroll
  for (int i = 0; i < 4; ++i) { r.u[i] = p0[i]; r.u[4 + i] = p1[i]; }
  return r.v;
}

// B (32x16 f16) from weight stored [N,K] row-major: lane col = col0+(lane&15),
// K = half*16 + {0..15} -> 8 contiguous dwords.
__device__ inline v16h frag_b_nk(const half_t* W, int ldb, int col0, int kb) {
  int lane = threadIdx.x & 31;
  const uint32_t* p = (const uint32_t*)(W + (size_t)(col0 + (lane & 15)) * ldb
                                        + kb + ((lane >> 4) << 4));
  FragH r;
#pragma unroll
  for (int i = 0; i < 8; ++i) r.u[i] = p[i];
  return r.v;
}

// B (32x16 f16) from K-major [K,N] array (e.g. V matrix in LDS).
__device__ inline v16h frag_b_kn(const half_t* Vm, int ldn, int col0, int kb) {
  int lane = threadIdx.x & 31;
  int c  = col0 + (lane & 15);
  int k0 = kb + ((lane >> 4) << 4);
  FragH r;
#pragma unroll
  for (int j = 0; j < 8; ++j) {
    r.h[2*j]   = Vm[(size_t)(k0 + 2*j)     * ldn + c];
    r.h[2*j+1] = Vm[(size_t)(k0 + 2*j + 1) * ldn + c];
  }
  return r.v;
}

// A loader with spatial (dy,dx) shift and zero padding (implicit-GEMM conv).
__device__ inline v16h frag_a_conv(const half_t* A, int row0, int kb, int dy, int dx) {
  int lane = threadIdx.x & 31;
  int r = row0 + (lane & 15);
  int b = r >> 12, p = r & 4095;
  int y = (p >> 6) + dy, x = (p & 63) + dx;
  FragH res;
  if ((unsigned)y < 64u && (unsigned)x < 64u) {
    const half_t* q = A + ((size_t)(b << 12) + (y << 6) + x) * DIM_
                        + kb + ((lane >> 4) << 4);
    const uint32_t* p0 = (const uint32_t*)q;
    const uint32_t* p1 = (const uint32_t*)(q + 8);
#pragma unroll
    for (int i = 0; i < 4; ++i) { res.u[i] = p0[i]; res.u[4+i] = p1[i]; }
  } else {
#pragma unroll
    for (int i = 0; i < 8; ++i) res.u[i] = 0u;
  }
  return res.v;
}

// ---------------- Generic WMMA GEMM: C[M,N] = A[M,K] @ W[N,K]^T + bias -------
// Each wave computes a 32(M) x 64(N) strip: 2 A fragments x 4 B fragments ->
// 8 accumulators, 8 wmma per 12 b128 loads per K step.
enum { OUT_F16 = 0, OUT_F16_GELU = 1, OUT_ADD_WINREV = 2, OUT_ADD_ID = 3, OUT_ADD_TRANS = 4 };

template<int AMODE, int OMODE, int KK>
__global__ __launch_bounds__(256) void gemm_wmma(
    const half_t* __restrict__ A, const half_t* __restrict__ W,
    const float* __restrict__ bias, void* __restrict__ Out,
    int M, int N, int shifted, int dy, int dx)
{
  int wave = threadIdx.x >> 5;
  int lane = threadIdx.x & 31;
  int stripsN = N >> 6;                       // strips of 64 columns
  int strip = blockIdx.x * 8 + wave;
  if (strip >= (M >> 5) * stripsN) return;
  int row0 = (strip / stripsN) << 5;          // 32 rows per strip
  int col0 = (strip % stripsN) << 6;

  v8f acc[2][4] = {};
#pragma unroll
  for (int kb = 0; kb < KK; kb += 32) {
    v16h a0, a1;
    if (AMODE == 0) {
      a0 = frag_a_rm(A, KK, row0, kb);
      a1 = frag_a_rm(A, KK, row0 + 16, kb);
    } else {
      a0 = frag_a_conv(A, row0, kb, dy, dx);
      a1 = frag_a_conv(A, row0 + 16, kb, dy, dx);
    }
#pragma unroll
    for (int j = 0; j < 4; ++j) {
      v16h b = frag_b_nk(W, KK, col0 + (j << 4), kb);
      acc[0][j] = __builtin_amdgcn_wmma_f32_16x16x32_f16(false, a0, false, b,
                                                         (short)0, acc[0][j], false, false);
      acc[1][j] = __builtin_amdgcn_wmma_f32_16x16x32_f16(false, a1, false, b,
                                                         (short)0, acc[1][j], false, false);
    }
  }

#pragma unroll
  for (int mi = 0; mi < 2; ++mi) {
    int rbase = row0 + (mi << 4) + ((lane >> 4) << 3);
#pragma unroll
    for (int j = 0; j < 4; ++j) {
      FragF c; c.v = acc[mi][j];
      int col = col0 + (j << 4) + (lane & 15);
      float bv = bias ? bias[col] : 0.f;
#pragma unroll
      for (int i = 0; i < 8; ++i) {
        int row = rbase + i;
        float v = c.f[i] + bv;
        if (OMODE == OUT_F16) {
          ((half_t*)Out)[(size_t)row * N + col] = (half_t)v;
        } else if (OMODE == OUT_F16_GELU) {
          float g = 0.5f * v * (1.f + erff(v * 0.70710678118654752f));
          ((half_t*)Out)[(size_t)row * N + col] = (half_t)g;
        } else if (OMODE == OUT_ADD_ID) {
          ((float*)Out)[(size_t)row * N + col] += v;
        } else if (OMODE == OUT_ADD_WINREV) {
          int w = row >> 6, n = row & 63;
          int b = w >> 6, wi = w & 63;
          int s = shifted ? SHIFT_ : 0;
          int yy = (((wi >> 3) << 3) + (n >> 3) + s) & 63;
          int xx = (((wi & 7) << 3) + (n & 7) + s) & 63;
          ((float*)Out)[((size_t)(b << 12) + (yy << 6) + xx) * DIM_ + col] += v;
        } else { // OUT_ADD_TRANS: out[b, col, p] += v
          int b = row >> 12, p = row & 4095;
          ((float*)Out)[(size_t)b * DIM_ * HW_ + (size_t)col * HW_ + p] += v;
        }
      }
    }
  }
}

// ---------------- LayerNorm family (one wave per token row) ------------------
__device__ inline float wred(float v) {
#pragma unroll
  for (int o = 16; o > 0; o >>= 1) v += __shfl_xor(v, o, 32);
  return v;
}

// x[B,DIM,HW] -> y[B*HW,DIM] with LN
__global__ __launch_bounds__(256) void embed_ln(const float* __restrict__ x,
    const float* __restrict__ g, const float* __restrict__ bt, float* __restrict__ y)
{
  int lane = threadIdx.x & 31;
  int row = blockIdx.x * 8 + (threadIdx.x >> 5);
  int b = row >> 12, p = row & 4095;
  const float* src = x + (size_t)b * DIM_ * HW_ + p;
  float v[6];
#pragma unroll
  for (int t = 0; t < 6; ++t) v[t] = src[(size_t)(lane + 32 * t) * HW_];
  float s = 0.f;
#pragma unroll
  for (int t = 0; t < 6; ++t) s += v[t];
  float mu = wred(s) * (1.f / 192.f);
  float q = 0.f;
#pragma unroll
  for (int t = 0; t < 6; ++t) { float d = v[t] - mu; q += d * d; }
  float inv = rsqrtf(wred(q) * (1.f / 192.f) + 1e-5f);
  float* dst = y + (size_t)row * DIM_;
#pragma unroll
  for (int t = 0; t < 6; ++t) {
    int c = lane + 32 * t;
    dst[c] = (v[t] - mu) * inv * g[c] + bt[c];
  }
}

// LN + (shift-roll) + window partition -> f16 rows in window order
__global__ __launch_bounds__(256) void ln_part(const float* __restrict__ y,
    const float* __restrict__ g, const float* __restrict__ bt,
    half_t* __restrict__ xw, int shifted)
{
  int lane = threadIdx.x & 31;
  int row = blockIdx.x * 8 + (threadIdx.x >> 5); // window-order row
  int w = row >> 6, n = row & 63;
  int b = w >> 6, wi = w & 63;
  int s = shifted ? SHIFT_ : 0;
  int yy = (((wi >> 3) << 3) + (n >> 3) + s) & 63;
  int xx = (((wi & 7) << 3) + (n & 7) + s) & 63;
  const float* src = y + ((size_t)(b << 12) + (yy << 6) + xx) * DIM_;
  float v[6];
#pragma unroll
  for (int t = 0; t < 6; ++t) v[t] = src[lane + 32 * t];
  float ssum = 0.f;
#pragma unroll
  for (int t = 0; t < 6; ++t) ssum += v[t];
  float mu = wred(ssum) * (1.f / 192.f);
  float q = 0.f;
#pragma unroll
  for (int t = 0; t < 6; ++t) { float d = v[t] - mu; q += d * d; }
  float inv = rsqrtf(wred(q) * (1.f / 192.f) + 1e-5f);
  half_t* dst = xw + (size_t)row * DIM_;
#pragma unroll
  for (int t = 0; t < 6; ++t) {
    int c = lane + 32 * t;
    dst[c] = (half_t)((v[t] - mu) * inv * g[c] + bt[c]);
  }
}

// plain LN -> f16, identity row order
__global__ __launch_bounds__(256) void ln_f16(const float* __restrict__ y,
    const float* __restrict__ g, const float* __restrict__ bt, half_t* __restrict__ o)
{
  int lane = threadIdx.x & 31;
  int row = blockIdx.x * 8 + (threadIdx.x >> 5);
  const float* src = y + (size_t)row * DIM_;
  float v[6];
#pragma unroll
  for (int t = 0; t < 6; ++t) v[t] = src[lane + 32 * t];
  float ssum = 0.f;
#pragma unroll
  for (int t = 0; t < 6; ++t) ssum += v[t];
  float mu = wred(ssum) * (1.f / 192.f);
  float q = 0.f;
#pragma unroll
  for (int t = 0; t < 6; ++t) { float d = v[t] - mu; q += d * d; }
  float inv = rsqrtf(wred(q) * (1.f / 192.f) + 1e-5f);
  half_t* dst = o + (size_t)row * DIM_;
#pragma unroll
  for (int t = 0; t < 6; ++t) {
    int c = lane + 32 * t;
    dst[c] = (half_t)((v[t] - mu) * inv * g[c] + bt[c]);
  }
}

// final LN -> zh f16 [row,c] AND initialize d_out f32 [b,c,p] (conv residual base)
__global__ __launch_bounds__(256) void ln_final(const float* __restrict__ y,
    const float* __restrict__ g, const float* __restrict__ bt,
    half_t* __restrict__ zh, float* __restrict__ out)
{
  int lane = threadIdx.x & 31;
  int row = blockIdx.x * 8 + (threadIdx.x >> 5);
  int b = row >> 12, p = row & 4095;
  const float* src = y + (size_t)row * DIM_;
  float v[6];
#pragma unroll
  for (int t = 0; t < 6; ++t) v[t] = src[lane + 32 * t];
  float ssum = 0.f;
#pragma unroll
  for (int t = 0; t < 6; ++t) ssum += v[t];
  float mu = wred(ssum) * (1.f / 192.f);
  float q = 0.f;
#pragma unroll
  for (int t = 0; t < 6; ++t) { float d = v[t] - mu; q += d * d; }
  float inv = rsqrtf(wred(q) * (1.f / 192.f) + 1e-5f);
#pragma unroll
  for (int t = 0; t < 6; ++t) {
    int c = lane + 32 * t;
    float val = (v[t] - mu) * inv * g[c] + bt[c];
    zh[(size_t)row * DIM_ + c] = (half_t)val;
    out[(size_t)b * DIM_ * HW_ + (size_t)c * HW_ + p] = val;
  }
}

// ---------------- Windowed attention (one WG per window x head) --------------
__device__ inline int regid3(int u) { return u < 56 ? 0 : (u < 60 ? 1 : 2); }

template<int SHIFTED>
__global__ __launch_bounds__(128) void attn_kernel(const half_t* __restrict__ qkv,
    const float* __restrict__ rpb, half_t* __restrict__ outw)
{
  int wh = blockIdx.x;             // 0 .. 1024*6-1
  int w = wh / HEADS_, h = wh % HEADS_;
  int tid = threadIdx.x, wave = tid >> 5, lane = tid & 31;

  __shared__ half_t qs[64 * 32], ks2[64 * 32], vs[64 * 32];
  __shared__ float  rpbS[225];
  __shared__ float  S[64 * 64];
  __shared__ half_t Ph[64 * 64];

  const half_t* base = qkv + (size_t)w * 64 * 576 + h * 32;
  for (int i = tid; i < 64 * 32; i += 128) {
    int n = i >> 5, d = i & 31;
    qs[i]  = base[(size_t)n * 576 + d];
    ks2[i] = base[(size_t)n * 576 + 192 + d];
    vs[i]  = base[(size_t)n * 576 + 384 + d];
  }
  for (int i = tid; i < 225; i += 128) rpbS[i] = rpb[i * HEADS_ + h];
  __syncthreads();

  // S = scale * q @ k^T + rel-pos bias (+ shift mask). 4x4 tiles, tn = wave.
  {
    int tn = wave;
    v16h bf = frag_b_nk(ks2, 32, tn * 16, 0);
    int wi = w & 63;
    int wy0 = (wi >> 3) << 3, wx0 = (wi & 7) << 3;
    int mcol = tn * 16 + (lane & 15);
    int my = mcol >> 3, mx = mcol & 7;
    int idm = SHIFTED ? (regid3(wy0 + my) * 3 + regid3(wx0 + mx)) : 0;
#pragma unroll
    for (int tm = 0; tm < 4; ++tm) {
      v16h a = frag_a_rm(qs, 32, tm * 16, 0);
      v8f acc = {};
      acc = __builtin_amdgcn_wmma_f32_16x16x32_f16(false, a, false, bf,
                                                   (short)0, acc, false, false);
      FragF c; c.v = acc;
      int rbase = tm * 16 + ((lane >> 4) << 3);
#pragma unroll
      for (int j = 0; j < 8; ++j) {
        int nrow = rbase + j;
        int ny = nrow >> 3, nx = nrow & 7;
        int rel = (ny - my + 7) * 15 + (nx - mx + 7);
        float v = c.f[j] * ATT_SCALE + rpbS[rel];
        if (SHIFTED) {
          int idn = regid3(wy0 + ny) * 3 + regid3(wx0 + nx);
          v += (idn != idm) ? -100.f : 0.f;
        }
        S[nrow * 64 + mcol] = v;
      }
    }
  }
  __syncthreads();

  // softmax rows (f32), write P as f16
  if (tid < 64) {
    float* r = S + tid * 64;
    float mx = -1e30f;
#pragma unroll 8
    for (int j = 0; j < 64; ++j) mx = fmaxf(mx, r[j]);
    float sum = 0.f;
#pragma unroll 8
    for (int j = 0; j < 64; ++j) { float e = __expf(r[j] - mx); r[j] = e; sum += e; }
    float inv = 1.f / sum;
    half_t* ph = Ph + tid * 64;
#pragma unroll 8
    for (int j = 0; j < 64; ++j) ph[j] = (half_t)(r[j] * inv);
  }
  __syncthreads();

  // O = P @ v : 4x2 tiles, 2 per wave, K=64
#pragma unroll
  for (int i = 0; i < 2; ++i) {
    int id = wave * 2 + i, tm = id >> 1, tn = id & 1;
    v8f acc = {};
#pragma unroll
    for (int kb = 0; kb < 64; kb += 32) {
      v16h a  = frag_a_rm(Ph, 64, tm * 16, kb);
      v16h bf = frag_b_kn(vs, 32, tn * 16, kb);
      acc = __builtin_amdgcn_wmma_f32_16x16x32_f16(false, a, false, bf,
                                                   (short)0, acc, false, false);
    }
    FragF c; c.v = acc;
    int d = tn * 16 + (lane & 15);
    int rb = tm * 16 + ((lane >> 4) << 3);
#pragma unroll
    for (int j = 0; j < 8; ++j) {
      int n = rb + j;
      outw[((size_t)w * 64 + n) * DIM_ + h * 32 + d] = (half_t)c.f[j];
    }
  }
}

// ---------------- weight conversion ------------------------------------------
__global__ void f32_to_f16(const float* __restrict__ s, half_t* __restrict__ d, int n) {
  int i = blockIdx.x * 256 + threadIdx.x;
  if (i < n) d[i] = (half_t)s[i];
}
// conv_w [C,IC,3,3] -> per-tap [tap][C][IC] f16
__global__ void conv_reorder(const float* __restrict__ s, half_t* __restrict__ d) {
  int i = blockIdx.x * 256 + threadIdx.x;
  if (i < 192 * 192 * 9) {
    int tap = i / (192 * 192), rem = i % (192 * 192);
    int c = rem / 192, ic = rem % 192;
    d[i] = (half_t)s[(size_t)(c * 192 + ic) * 9 + tap];
  }
}

// ---------------- host orchestration -----------------------------------------
extern "C" void kernel_launch(void* const* d_in, const int* in_sizes, int n_in,
                              void* d_out, int out_size, void* d_ws, size_t ws_size,
                              hipStream_t stream) {
  (void)in_sizes; (void)n_in; (void)out_size; (void)ws_size;
  char* ws = (char*)d_ws;
  float*  y   = (float*)ws;                          // 65536*192 f32 = 50.3 MB
  size_t  off = (size_t)MROWS_ * DIM_ * sizeof(float);
  half_t* xw  = (half_t*)(ws + off);                 // 65536*192 f16 (reused)
  off += (size_t)MROWS_ * DIM_ * sizeof(half_t);
  half_t* tA  = (half_t*)(ws + off);                 // qkv (576) / mlp hidden (768) f16
  off += (size_t)MROWS_ * 768 * sizeof(half_t);
  half_t* qkv_wh  = (half_t*)(ws + off);
  half_t* proj_wh = qkv_wh  + 6 * 576 * 192;
  half_t* fc1_wh  = proj_wh + 6 * 192 * 192;
  half_t* fc2_wh  = fc1_wh  + 6 * 768 * 192;
  half_t* conv_wh = fc2_wh  + 6 * 192 * 768;

  const float* x    = (const float*)d_in[0];
  const float* en_g = (const float*)d_in[1];
  const float* en_b = (const float*)d_in[2];

  // weight down-conversion (same result each call -> graph-safe)
  auto cvt = [&](const void* s, half_t* d, int n) {
    f32_to_f16<<<(n + 255) / 256, 256, 0, stream>>>((const float*)s, d, n);
  };
  cvt(d_in[5],  qkv_wh, 6 * 576 * 192);
  cvt(d_in[7],  proj_wh, 6 * 192 * 192);
  cvt(d_in[12], fc1_wh, 6 * 768 * 192);
  cvt(d_in[14], fc2_wh, 6 * 192 * 768);
  conv_reorder<<<(192 * 192 * 9 + 255) / 256, 256, 0, stream>>>((const float*)d_in[18], conv_wh);

  // patch embed + LN -> y
  embed_ln<<<MROWS_ / 8, 256, 0, stream>>>(x, en_g, en_b, y);

  for (int blk = 0; blk < 6; ++blk) {
    int shifted = blk & 1;
    const float* n1g = (const float*)d_in[3]  + blk * 192;
    const float* n1b = (const float*)d_in[4]  + blk * 192;
    const float* qkb = (const float*)d_in[6]  + blk * 576;
    const float* pjb = (const float*)d_in[8]  + blk * 192;
    const float* rpb = (const float*)d_in[9]  + blk * 225 * 6;
    const float* n2g = (const float*)d_in[10] + blk * 192;
    const float* n2b = (const float*)d_in[11] + blk * 192;
    const float* f1b = (const float*)d_in[13] + blk * 768;
    const float* f2b = (const float*)d_in[15] + blk * 192;

    // LN1 + roll + window partition (f16)
    ln_part<<<MROWS_ / 8, 256, 0, stream>>>(y, n1g, n1b, xw, shifted);
    // QKV: [65536,192] x [576,192]^T -> tA f16   (strips: 2048 * 9 / 8)
    gemm_wmma<0, OUT_F16, 192><<<(MROWS_ / 32) * (576 / 64) / 8, 256, 0, stream>>>(
        xw, qkv_wh + (size_t)blk * 576 * 192, qkb, tA, MROWS_, 576, 0, 0, 0);
    // attention -> xw (window-order, [.,64,192])
    if (shifted)
      attn_kernel<1><<<1024 * HEADS_, 128, 0, stream>>>(tA, rpb, xw);
    else
      attn_kernel<0><<<1024 * HEADS_, 128, 0, stream>>>(tA, rpb, xw);
    // proj + window-reverse + unroll + residual add into y
    gemm_wmma<0, OUT_ADD_WINREV, 192><<<(MROWS_ / 32) * (192 / 64) / 8, 256, 0, stream>>>(
        xw, proj_wh + (size_t)blk * 192 * 192, pjb, y, MROWS_, 192, shifted, 0, 0);
    // LN2 -> f16
    ln_f16<<<MROWS_ / 8, 256, 0, stream>>>(y, n2g, n2b, xw);
    // fc1 + exact GELU -> tA f16
    gemm_wmma<0, OUT_F16_GELU, 192><<<(MROWS_ / 32) * (768 / 64) / 8, 256, 0, stream>>>(
        xw, fc1_wh + (size_t)blk * 768 * 192, f1b, tA, MROWS_, 768, 0, 0, 0);
    // fc2 + residual add into y
    gemm_wmma<0, OUT_ADD_ID, 768><<<(MROWS_ / 32) * (192 / 64) / 8, 256, 0, stream>>>(
        tA, fc2_wh + (size_t)blk * 192 * 768, f2b, y, MROWS_, 192, 0, 0, 0);
  }

  // final LN: zh f16 for conv input, and initialize d_out with the residual base
  const float* fng = (const float*)d_in[16];
  const float* fnb = (const float*)d_in[17];
  ln_final<<<MROWS_ / 8, 256, 0, stream>>>(y, fng, fnb, xw, (float*)d_out);

  // 3x3 conv as 9 shifted implicit GEMMs accumulating into d_out
  for (int ky = 0; ky < 3; ++ky)
    for (int kx = 0; kx < 3; ++kx)
      gemm_wmma<1, OUT_ADD_TRANS, 192><<<(MROWS_ / 32) * (192 / 64) / 8, 256, 0, stream>>>(
          xw, conv_wh + (size_t)(ky * 3 + kx) * 192 * 192, nullptr, d_out,
          MROWS_, 192, 0, ky - 1, kx - 1);
}
